// LinearMixtralSparseMoeBlock_5729486373375
// MI455X (gfx1250) — matrix operand order repaired and checked by reading, not compiled, not run on hardware
//
#include <hip/hip_runtime.h>
#include <hip/hip_bf16.h>

#define T_TOK 4096
#define HDIM  1024
#define FDIM  3584
#define NEXP  8

typedef __bf16 bf16;
typedef __attribute__((ext_vector_type(16))) __bf16 v16bf;
typedef __attribute__((ext_vector_type(8)))  float  v8f;
typedef __attribute__((ext_vector_type(4)))  unsigned u32x4;
typedef __attribute__((ext_vector_type(8)))  int      i32x8;
typedef __attribute__((ext_vector_type(4)))  int      i32x4;

// ---------- helpers ----------
__device__ inline bf16 f32_to_bf16(float f) { return (bf16)f; }

// native pack: two f32 -> one dword of 2 bf16 (v_cvt_pk_bf16_f32 path)
__device__ inline unsigned pack2_bf16(float lo, float hi) {
  union { bf16 h[2]; unsigned u; } p;
  p.h[0] = (bf16)lo;
  p.h[1] = (bf16)hi;
  return p.u;
}

__device__ inline v8f wmma_bf16(v16bf a, v16bf b, v8f c) {
  return __builtin_amdgcn_wmma_f32_16x16x32_bf16(false, a, false, b, (short)0, c,
                                                 false, false);
}

// CDNA5 async global->LDS copy (ASYNCcnt tracked), per-lane 16B
__device__ inline void async_copy_b128(unsigned ldsaddr, unsigned long long gaddr) {
  asm volatile("global_load_async_to_lds_b128 %0, %1, off"
               :: "v"(ldsaddr), "v"(gaddr) : "memory");
}
__device__ inline void wait_asynccnt0() {
  asm volatile("s_wait_asynccnt 0" ::: "memory");
}

// A fragment: 16x32 bf16 tile, row-major in LDS with row stride 40 elements.
__device__ inline v16bf load_frag_a(const bf16* sA, int lane) {
  const int m = lane & 15, half = lane >> 4;
  union { v16bf v; unsigned u[8]; } a;
#pragma unroll
  for (int v = 0; v < 8; ++v) {
    const int kb = 2 * v + 16 * half;
    a.u[v] = *(const unsigned*)&sA[m * 40 + kb];
  }
  return a.v;
}

// B fragment from LDS stored TRANSPOSED: sBT[n][k], row stride 40 elements.
__device__ inline v16bf load_frag_bt(const bf16* sBT, int lane) {
  const int n = lane & 15, half = lane >> 4;
  union { v16bf v; uint4 q[2]; } b;
  const uint4* p = (const uint4*)&sBT[n * 40 + half * 16];
  b.q[0] = p[0];
  b.q[1] = p[1];
  return b.v;
}

// ---------- small kernels ----------
__global__ void zero_f32_kernel(float* __restrict__ p, int n) {
  for (int i = blockIdx.x * blockDim.x + threadIdx.x; i < n; i += gridDim.x * blockDim.x)
    p[i] = 0.f;
}
__global__ void zero_i32_kernel(int* __restrict__ p, int n) {
  int i = threadIdx.x;
  if (i < n) p[i] = 0;
}
__global__ void cast_bf16_kernel(const float* __restrict__ x, bf16* __restrict__ y, int n) {
  for (int i = blockIdx.x * blockDim.x + threadIdx.x; i < n; i += gridDim.x * blockDim.x)
    y[i] = (bf16)x[i];
}

// one block (8 waves) per token; wave w computes logits for expert w
__global__ __launch_bounds__(256) void router_kernel(const float* __restrict__ x,
                                                     const float* __restrict__ rw,
                                                     float* __restrict__ combine) {
  const int t = blockIdx.x;
  const int lane = threadIdx.x & 31, w = threadIdx.x >> 5;
  const float* xr = x + (size_t)t * HDIM;
  const float* rr = rw + (size_t)w * HDIM;
  float p = 0.f;
  for (int h = lane; h < HDIM; h += 32) p += xr[h] * rr[h];
#pragma unroll
  for (int o = 16; o > 0; o >>= 1) p += __shfl_down(p, o, 32);
  __shared__ float slog[NEXP];
  if (lane == 0) slog[w] = p;
  __syncthreads();
  if (threadIdx.x == 0) {
    float mx = slog[0];
#pragma unroll
    for (int e = 1; e < NEXP; ++e) mx = fmaxf(mx, slog[e]);
    float ex[NEXP];
#pragma unroll
    for (int e = 0; e < NEXP; ++e) ex[e] = __expf(slog[e] - mx);
    int i1 = 0;
#pragma unroll
    for (int e = 1; e < NEXP; ++e) if (ex[e] > ex[i1]) i1 = e;
    int i2 = (i1 == 0) ? 1 : 0;
#pragma unroll
    for (int e = 0; e < NEXP; ++e) if (e != i1 && ex[e] > ex[i2]) i2 = e;
    const float inv = 1.f / (ex[i1] + ex[i2]);
#pragma unroll
    for (int e = 0; e < NEXP; ++e)
      combine[(size_t)t * NEXP + e] =
          (e == i1) ? ex[i1] * inv : ((e == i2) ? ex[i2] * inv : 0.f);
  }
}

__global__ void build_lists_kernel(const float* __restrict__ combine,
                                   int* __restrict__ counts, int* __restrict__ lists) {
  const int t = blockIdx.x * blockDim.x + threadIdx.x;
  if (t >= T_TOK) return;
#pragma unroll
  for (int e = 0; e < NEXP; ++e) {
    if (combine[(size_t)t * NEXP + e] > 0.f) {
      int p = atomicAdd(&counts[e], 1);
      lists[(size_t)e * T_TOK + p] = t;
    }
  }
}

// ---------- gate/up fused GEMM: act = silu(x@Wg) * (x@Wu) on gathered rows ----------
// block tile 64(M) x 128(N); 8 waves as 2(M) x 4(N), each wave 32x32 (2x2 WMMA) x2 mats
__global__ __launch_bounds__(256) void gateup_kernel(
    const bf16* __restrict__ xbf, const float* __restrict__ wg,
    const float* __restrict__ wu, const int* __restrict__ counts,
    const int* __restrict__ lists, bf16* __restrict__ act, int e) {
  const int cnt = counts[e];
  const int m0 = blockIdx.y * 64;
  if (m0 >= cnt) return;
  const int n0 = blockIdx.x * 128;
  const int tid = threadIdx.x;
  const int lane = tid & 31, w = tid >> 5, wm = w >> 2, wn = w & 3;

  __shared__ bf16 sA[64 * 40];        // 64 rows(m) x 32(k), stride 40
  __shared__ bf16 sBgT[128 * 40];     // TRANSPOSED: 128 rows(n) x 32(k), stride 40
  __shared__ bf16 sBuT[128 * 40];

  const float* wge = wg + (size_t)e * HDIM * FDIM;
  const float* wue = wu + (size_t)e * HDIM * FDIM;
  const int* lst = lists + (size_t)e * T_TOK;

  // A loading: thread -> (row 0..63, 8-element segment 0..3); async copy to LDS
  const int arow = tid >> 2;
  const int aseg = (tid & 3) * 8;
  const int grow_a = m0 + arow;
  const int tok = (grow_a < cnt) ? lst[grow_a] : lst[m0];
  const uint4* asrc = (const uint4*)(xbf + (size_t)tok * HDIM + aseg);
  const unsigned lds_a = (unsigned)(size_t)&sA[arow * 40 + aseg];

  // B loading (transposing): thread -> column n = tid%128, k-group kg = tid/128
  const int bn = tid & 127;
  const int bkg = (tid >> 7) * 16;    // 0 or 16

  const v8f vz = {0.f, 0.f, 0.f, 0.f, 0.f, 0.f, 0.f, 0.f};
  v8f accg[2][2], accu[2][2];
#pragma unroll
  for (int i = 0; i < 2; ++i)
#pragma unroll
    for (int j = 0; j < 2; ++j) { accg[i][j] = vz; accu[i][j] = vz; }

  for (int kt = 0; kt < HDIM / 32; ++kt) {
    const int k0 = kt * 32;
    // stage A (64x32 bf16) via CDNA5 async global->LDS (ASYNCcnt)
    async_copy_b128(lds_a, (unsigned long long)(size_t)(asrc + kt * 4));
    // stage B gate & up, transposed: coalesced dword reads, native bf16 pack
    {
      const float* gsrc = wge + (size_t)(k0 + bkg) * FDIM + n0 + bn;
      const float* usrc = wue + (size_t)(k0 + bkg) * FDIM + n0 + bn;
      unsigned pg[8], pu[8];
#pragma unroll
      for (int q = 0; q < 8; ++q) {
        pg[q] = pack2_bf16(gsrc[(size_t)(2 * q) * FDIM], gsrc[(size_t)(2 * q + 1) * FDIM]);
        pu[q] = pack2_bf16(usrc[(size_t)(2 * q) * FDIM], usrc[(size_t)(2 * q + 1) * FDIM]);
      }
      *(uint4*)&sBgT[bn * 40 + bkg]     = make_uint4(pg[0], pg[1], pg[2], pg[3]);
      *(uint4*)&sBgT[bn * 40 + bkg + 8] = make_uint4(pg[4], pg[5], pg[6], pg[7]);
      *(uint4*)&sBuT[bn * 40 + bkg]     = make_uint4(pu[0], pu[1], pu[2], pu[3]);
      *(uint4*)&sBuT[bn * 40 + bkg + 8] = make_uint4(pu[4], pu[5], pu[6], pu[7]);
    }
    if (kt + 1 < HDIM / 32) {
      __builtin_prefetch(wge + (size_t)(k0 + 32 + bkg) * FDIM + n0 + bn, 0, 0);
      __builtin_prefetch(wue + (size_t)(k0 + 32 + bkg) * FDIM + n0 + bn, 0, 0);
    }
    wait_asynccnt0();                 // LDS A visible before barrier
    __syncthreads();

    const v16bf a0 = load_frag_a(&sA[(wm * 32 + 0) * 40], lane);
    const v16bf a1 = load_frag_a(&sA[(wm * 32 + 16) * 40], lane);
#pragma unroll
    for (int j = 0; j < 2; ++j) {
      const v16bf bg = load_frag_bt(&sBgT[(wn * 32 + j * 16) * 40], lane);
      const v16bf bu = load_frag_bt(&sBuT[(wn * 32 + j * 16) * 40], lane);
      accg[0][j] = wmma_bf16(a0, bg, accg[0][j]);
      accg[1][j] = wmma_bf16(a1, bg, accg[1][j]);
      accu[0][j] = wmma_bf16(a0, bu, accu[0][j]);
      accu[1][j] = wmma_bf16(a1, bu, accu[1][j]);
    }
    __syncthreads();
  }

  // epilogue: act = silu(g) * u  (bf16, indexed by list position)
  const int half = lane >> 4, nloc = lane & 15;
#pragma unroll
  for (int i = 0; i < 2; ++i)
#pragma unroll
    for (int j = 0; j < 2; ++j) {
#pragma unroll
      for (int v = 0; v < 8; ++v) {
        const int row = wm * 32 + i * 16 + v + 8 * half;
        const int grow = m0 + row;
        if (grow < cnt) {
          const float g = accg[i][j][v];
          const float u = accu[i][j][v];
          const float s = g / (1.f + __expf(-g));
          const int col = n0 + wn * 32 + j * 16 + nloc;
          act[(size_t)grow * FDIM + col] = f32_to_bf16(s * u);
        }
      }
    }
}

// ---------- down GEMM + weighted scatter-add: out[token] += combine * (act @ Wd) ----------
// block tile 64(M) x 256(N); 8 waves as 2(M) x 4(N), each wave 32x64 (2x4 WMMA)
// A tile staged by the Tensor Data Mover (TDM) with LDS padding reproducing stride 40.
__global__ __launch_bounds__(256) void down_kernel(
    const bf16* __restrict__ act, const float* __restrict__ wd,
    const int* __restrict__ counts, const int* __restrict__ lists,
    const float* __restrict__ combine, float* __restrict__ out, int e) {
  const int cnt = counts[e];
  const int m0 = blockIdx.y * 64;
  if (m0 >= cnt) return;
  const int n0 = blockIdx.x * 256;
  const int tid = threadIdx.x;
  const int lane = tid & 31, w = tid >> 5, wm = w >> 2, wn = w & 3;

  __shared__ bf16 sA[64 * 40];
  __shared__ bf16 sBT[256 * 40];      // TRANSPOSED: 256 rows(n) x 32(k), stride 40

  const float* wde = wd + (size_t)e * FDIM * HDIM;
  const int* lst = lists + (size_t)e * T_TOK;

#if __has_builtin(__builtin_amdgcn_tensor_load_to_lds)
  // TDM descriptor group1 (constant across k-tiles):
  // data_size=2B, pad_enable, pad_interval=16 dwords (one 32-elem row),
  // pad_amount=4 dwords (8 elems) -> LDS row stride 40 elements.
  i32x8 g1;
  g1[0] = (1 << 16) | (1 << 20) | (3 << 22) | (3 << 25);
  g1[1] = 0;                       // abar=0, tensor_dim0[15:0]=0
  g1[2] = 0x4000;                  // tensor_dim0 = 0x40000000 (no OOB), dim1 lo = 0
  g1[3] = 0x4000 | (32 << 16);     // tensor_dim1 = 0x40000000, tile_dim0 = 32
  g1[4] = 64;                      // tile_dim1 = 64, tile_dim2 = 0
  g1[5] = FDIM;                    // tensor_dim0_stride = FDIM elements
  g1[6] = 0;
  g1[7] = 0;
  const i32x4 gz4 = {0, 0, 0, 0};
  const i32x8 gz8 = {0, 0, 0, 0, 0, 0, 0, 0};
  const unsigned lds_a_base = (unsigned)(size_t)&sA[0];
#else
  const int arow = tid >> 2;
  const int aseg = (tid & 3) * 8;
  const uint4* asrc = (const uint4*)(act + (size_t)(m0 + arow) * FDIM + aseg);
#endif

  // B loading (transposing): thread t owns column n = t, all 32 k's
  const int bn = tid;

  const v8f vz = {0.f, 0.f, 0.f, 0.f, 0.f, 0.f, 0.f, 0.f};
  v8f acc[2][4];
#pragma unroll
  for (int i = 0; i < 2; ++i)
#pragma unroll
    for (int j = 0; j < 4; ++j) acc[i][j] = vz;

  for (int kt = 0; kt < FDIM / 32; ++kt) {
    const int k0 = kt * 32;
#if __has_builtin(__builtin_amdgcn_tensor_load_to_lds)
    if (tid < 32) {                       // wave 0 drives the TDM
      const unsigned long long ga =
          (unsigned long long)(size_t)(act + (size_t)m0 * FDIM + k0);
      u32x4 g0;
      g0[0] = 1u;                          // count=1
      g0[1] = lds_a_base;                  // lds_addr
      g0[2] = (unsigned)(ga & 0xffffffffull);
      g0[3] = (unsigned)((ga >> 32) & 0x1ffffffull) | (2u << 30);  // type=2
      __builtin_amdgcn_tensor_load_to_lds(g0, g1, gz4, gz4, gz8, 0);
      __builtin_amdgcn_s_wait_tensorcnt(0);
    }
#else
    *(uint4*)&sA[arow * 40 + aseg] = asrc[kt * 4];
#endif
    {
      const float* bsrc = wde + (size_t)k0 * HDIM + n0 + bn;
      unsigned pb[16];
#pragma unroll
      for (int q = 0; q < 16; ++q)
        pb[q] = pack2_bf16(bsrc[(size_t)(2 * q) * HDIM], bsrc[(size_t)(2 * q + 1) * HDIM]);
      *(uint4*)&sBT[bn * 40 + 0]  = make_uint4(pb[0],  pb[1],  pb[2],  pb[3]);
      *(uint4*)&sBT[bn * 40 + 8]  = make_uint4(pb[4],  pb[5],  pb[6],  pb[7]);
      *(uint4*)&sBT[bn * 40 + 16] = make_uint4(pb[8],  pb[9],  pb[10], pb[11]);
      *(uint4*)&sBT[bn * 40 + 24] = make_uint4(pb[12], pb[13], pb[14], pb[15]);
    }
    if (kt + 1 < FDIM / 32)
      __builtin_prefetch(wde + (size_t)(k0 + 32) * HDIM + n0 + bn, 0, 0);
    __syncthreads();

    const v16bf a0 = load_frag_a(&sA[(wm * 32 + 0) * 40], lane);
    const v16bf a1 = load_frag_a(&sA[(wm * 32 + 16) * 40], lane);
#pragma unroll
    for (int j = 0; j < 4; ++j) {
      const v16bf b = load_frag_bt(&sBT[(wn * 64 + j * 16) * 40], lane);
      acc[0][j] = wmma_bf16(a0, b, acc[0][j]);
      acc[1][j] = wmma_bf16(a1, b, acc[1][j]);
    }
    __syncthreads();
  }

  const int half = lane >> 4, nloc = lane & 15;
#pragma unroll
  for (int i = 0; i < 2; ++i)
#pragma unroll
    for (int j = 0; j < 4; ++j) {
#pragma unroll
      for (int v = 0; v < 8; ++v) {
        const int row = wm * 32 + i * 16 + v + 8 * half;
        const int grow = m0 + row;
        if (grow < cnt) {
          const int token = lst[grow];
          const float cw = combine[(size_t)token * NEXP + e];
          const int col = n0 + wn * 64 + j * 16 + nloc;
          out[(size_t)token * HDIM + col] += cw * acc[i][j][v];
        }
      }
    }
}

// ---------- launcher ----------
extern "C" void kernel_launch(void* const* d_in, const int* in_sizes, int n_in,
                              void* d_out, int out_size, void* d_ws, size_t ws_size,
                              hipStream_t stream) {
  const float* x  = (const float*)d_in[0];
  const float* rw = (const float*)d_in[1];
  const float* wg = (const float*)d_in[2];
  const float* wu = (const float*)d_in[3];
  const float* wd = (const float*)d_in[4];
  float* out = (float*)d_out;

  char* ws = (char*)d_ws;
  size_t off = 0;
  bf16* xbf = (bf16*)(ws + off);      off += (size_t)T_TOK * HDIM * sizeof(bf16);
  bf16* act = (bf16*)(ws + off);      off += (size_t)T_TOK * FDIM * sizeof(bf16);
  float* combine = (float*)(ws + off); off += (size_t)T_TOK * NEXP * sizeof(float);
  int* counts = (int*)(ws + off);     off += 256;
  int* lists = (int*)(ws + off);      off += (size_t)NEXP * T_TOK * sizeof(int);
  (void)ws_size; (void)in_sizes; (void)n_in; (void)out_size;

  zero_f32_kernel<<<2048, 256, 0, stream>>>(out, T_TOK * HDIM);
  zero_i32_kernel<<<1, 32, 0, stream>>>(counts, NEXP);
  router_kernel<<<T_TOK, 256, 0, stream>>>(x, rw, combine);
  cast_bf16_kernel<<<2048, 256, 0, stream>>>(x, xbf, T_TOK * HDIM);
  build_lists_kernel<<<T_TOK / 256, 256, 0, stream>>>(combine, counts, lists);

  for (int e = 0; e < NEXP; ++e) {
    gateup_kernel<<<dim3(FDIM / 128, T_TOK / 64), 256, 0, stream>>>(
        xbf, wg, wu, counts, lists, act, e);
    down_kernel<<<dim3(HDIM / 256, T_TOK / 64), 256, 0, stream>>>(
        act, wd, counts, lists, combine, out, e);
  }
}
